// GNNSceneEmbeddingNetwork_LearnedEdgeVector_19018115186962
// MI455X (gfx1250) — compile-verified, matrix operand order checked by reading, not run on hardware
//
#include <hip/hip_runtime.h>
#include <hip/hip_bf16.h>

// ---------------- problem constants ----------------
#define NNODES 50000
#define MP     50048          // NNODES rounded up to 128 (GEMM tile M)
#define NEDGES 600000
#define DIN    518
#define KP1    544            // DIN rounded up to 32
#define DMODEL 128
#define EDIM   32
#define SDIM   32
#define NEG_SLOPE 0.2f

typedef __attribute__((ext_vector_type(16))) __bf16 v16bf;
typedef __attribute__((ext_vector_type(8)))  float  v8f;
typedef __attribute__((ext_vector_type(4)))  int    v4i;
typedef __attribute__((address_space(1))) v4i* gv4p;   // global int4*
typedef __attribute__((address_space(3))) v4i* lv4p;   // LDS int4*

#if __has_builtin(__builtin_amdgcn_global_load_async_to_lds_b128)
#define ASYNC_STAGE 1
#else
#define ASYNC_STAGE 0
#endif

__device__ __forceinline__ void wait_async0() {
#if __has_builtin(__builtin_amdgcn_s_wait_asynccnt)
  __builtin_amdgcn_s_wait_asynccnt(0);
#else
  asm volatile("s_wait_asynccnt 0x0" ::: "memory");
#endif
}

__device__ __forceinline__ __bf16 f2bf(float f) {
  unsigned u = __builtin_bit_cast(unsigned, f);
  u += 0x7FFFu + ((u >> 16) & 1u);              // round-to-nearest-even
  return __builtin_bit_cast(__bf16, (unsigned short)(u >> 16));
}
__device__ __forceinline__ __bf16 bf0() {
  return __builtin_bit_cast(__bf16, (unsigned short)0);
}

// monotone float<->uint mapping for atomicMax-based segment max
__device__ __forceinline__ unsigned fmapkey(float x) {
  unsigned u = __builtin_bit_cast(unsigned, x);
  return (u & 0x80000000u) ? ~u : (u | 0x80000000u);
}
__device__ __forceinline__ float funmap(unsigned k) {
  unsigned u = (k & 0x80000000u) ? (k & 0x7FFFFFFFu) : ~k;
  return __builtin_bit_cast(float, u);
}

// ---------------- cast / pad kernels ----------------
__global__ void k_castA(const float* __restrict__ A, __bf16* __restrict__ out,
                        int M, int K, int Kp) {
  int m = blockIdx.y;
  int k = blockIdx.x * blockDim.x + threadIdx.x;
  if (k >= Kp) return;
  out[(long)m * Kp + k] = (m < M && k < K) ? f2bf(A[(long)m * K + k]) : bf0();
}

// B [K,N] f32 -> out [N,Kp] bf16 (transposed, K zero-padded)
__global__ void k_castBT(const float* __restrict__ B, __bf16* __restrict__ out,
                         int K, int N, int Kp) {
  int n = blockIdx.y;
  int k = blockIdx.x * blockDim.x + threadIdx.x;
  if (k >= Kp) return;
  out[(long)n * Kp + k] = (k < K) ? f2bf(B[(long)k * N + n]) : bf0();
}

// ---------------- bf16 WMMA GEMM ----------------
// A: [Mp,Kp] bf16 padded row-major.  Bt: [Ncols,Kp] bf16 (pre-transposed).
// Mp % 128 == 0, Kp % 32 == 0, Ncols % 64 == 0 -> no guards anywhere.
#define BM 128
#define BN 64
#define BK 32
#define LDST 40   // padded LDS stride (bf16 elems); row*80B keeps 16B alignment

template <bool HAS_BIAS, bool RELU, bool OUTF, bool OUTB>
__global__ __launch_bounds__(256) void k_gemm_bf16wmma(
    const __bf16* __restrict__ A, const __bf16* __restrict__ Bt,
    const float* __restrict__ bias, float* __restrict__ Cf,
    __bf16* __restrict__ Cb, int Ncols, int Kp)
{
  __shared__ __bf16 As[BM * LDST];
  __shared__ __bf16 Bs[BN * LDST];

  const int tid  = threadIdx.x;
  const int lane = tid & 31, wave = tid >> 5;
  const int wm = wave >> 1, wn = wave & 1;       // 4x2 wave grid
  const int half = lane >> 4, l16 = lane & 15;
  const long rowBase = (long)blockIdx.x * BM;
  const int  colBase = blockIdx.y * BN;

  v8f acc[2][2];
#pragma unroll
  for (int i = 0; i < 2; i++)
#pragma unroll
    for (int j = 0; j < 2; j++) acc[i][j] = (v8f){0,0,0,0,0,0,0,0};

  // staging maps (all unconditional, 16B-aligned vector ops)
  const int arow  = tid >> 1;           // A: 128 rows x 32 k, 16 bf16 per thread
  const int akseg = (tid & 1) * 16;
  const int bcol  = tid >> 2;           // B: 64 cols x 32 k, 8 bf16 per thread
  const int bkseg = (tid & 3) * 8;

  const __bf16* aptr = A + (rowBase + arow) * (long)Kp + akseg;
  const __bf16* bptr = Bt + (long)(colBase + bcol) * Kp + bkseg;
  __bf16* asl = &As[arow * LDST + akseg];
  __bf16* bsl = &Bs[bcol * LDST + bkseg];

  for (int k0 = 0; k0 < Kp; k0 += BK) {
#if ASYNC_STAGE
    { // CDNA5 async copy: global -> LDS without VGPR round trip (ASYNCcnt)
      __builtin_amdgcn_global_load_async_to_lds_b128(
          (gv4p)(void*)const_cast<__bf16*>(aptr + k0),     (lv4p)(void*)asl,       0, 0);
      __builtin_amdgcn_global_load_async_to_lds_b128(
          (gv4p)(void*)const_cast<__bf16*>(aptr + k0 + 8), (lv4p)(void*)(asl + 8), 0, 0);
      __builtin_amdgcn_global_load_async_to_lds_b128(
          (gv4p)(void*)const_cast<__bf16*>(bptr + k0),     (lv4p)(void*)bsl,       0, 0);
      wait_async0();
    }
#else
    {
      uint4 va0 = *(const uint4*)(aptr + k0);
      uint4 va1 = *(const uint4*)(aptr + k0 + 8);
      uint4 vb  = *(const uint4*)(bptr + k0);
      *(uint4*)asl       = va0;
      *(uint4*)(asl + 8) = va1;
      *(uint4*)bsl       = vb;
    }
#endif
    __syncthreads();

    // fragment fill per ISA 16-bit A layout: lane=row(M), element e -> K
    v16bf afr[2], bfr[2];
#pragma unroll
    for (int sm = 0; sm < 2; sm++) {
      int row = wm * 32 + sm * 16 + l16;
#pragma unroll
      for (int e = 0; e < 16; e++) {
        int k = ((e >> 3) << 4) + half * 8 + (e & 7);
        afr[sm][e] = As[row * LDST + k];
      }
    }
#pragma unroll
    for (int sn = 0; sn < 2; sn++) {
      int col = wn * 32 + sn * 16 + l16;
#pragma unroll
      for (int e = 0; e < 16; e++) {
        int k = ((e >> 3) << 4) + half * 8 + (e & 7);
        bfr[sn][e] = Bs[col * LDST + k];
      }
    }
#pragma unroll
    for (int sm = 0; sm < 2; sm++)
#pragma unroll
      for (int sn = 0; sn < 2; sn++)
        acc[sm][sn] = __builtin_amdgcn_wmma_f32_16x16x32_bf16(
            false, afr[sm], false, bfr[sn], (short)0, acc[sm][sn], false, false);
    __syncthreads();
  }

  // epilogue: compile-time specialized, strided pointer walk (no per-elem mul)
#pragma unroll
  for (int sn = 0; sn < 2; sn++) {
    const int n = colBase + wn * 32 + sn * 16 + l16;
    const float bval = HAS_BIAS ? bias[n] : 0.0f;
#pragma unroll
    for (int sm = 0; sm < 2; sm++) {
      const long m0 = rowBase + wm * 32 + sm * 16 + half * 8;
      float*  cf = OUTF ? (Cf + m0 * Ncols + n) : nullptr;
      __bf16* cb = OUTB ? (Cb + m0 * Ncols + n) : nullptr;
#pragma unroll
      for (int r = 0; r < 8; r++) {
        float v = acc[sm][sn][r] + bval;
        if (RELU) v = v > 0.f ? v : 0.f;
        if (OUTF) cf[(long)r * Ncols] = v;
        if (OUTB) cb[(long)r * Ncols] = f2bf(v);
      }
    }
  }
}

// ---------------- elementwise / edge kernels ----------------
__global__ void k_zero(float* p, long n) {
  long i = (long)blockIdx.x * blockDim.x + threadIdx.x;
  if (i < n) p[i] = 0.0f;
}

__global__ void k_edge_embed(const int* __restrict__ eattr, const int* __restrict__ eidx,
                             const float* __restrict__ rel, float* __restrict__ e2,
                             float* __restrict__ eloop_sum, float* __restrict__ cnt) {
  int idx = blockIdx.x * blockDim.x + threadIdx.x;
  if (idx >= NEDGES * EDIM) return;
  int i = idx >> 5, d = idx & 31;
  float v = rel[eattr[i] * EDIM + d];
  e2[idx] = v;
  int dn = eidx[NEDGES + i];
  atomicAdd(&eloop_sum[dn * EDIM + d], v);
  if (d == 0) atomicAdd(&cnt[dn], 1.0f);
}

__global__ void k_eloop(const float* __restrict__ eloop_sum, const float* __restrict__ cnt,
                        float* __restrict__ e2) {
  int idx = blockIdx.x * blockDim.x + threadIdx.x;
  if (idx >= NNODES * EDIM) return;
  int n = idx >> 5;
  float c = cnt[n]; c = c < 1.0f ? 1.0f : c;
  e2[(long)(NEDGES + n) * EDIM + (idx & 31)] = eloop_sum[idx] / c;
}

// wa[j] = sum_d line[j][d] * a_edge[d]   (line: [EDIM, DMODEL])
__global__ void k_wa(const float* __restrict__ line, const float* __restrict__ ae,
                     float* __restrict__ wa) {
  int j = threadIdx.x;  // 32 threads
  float s = 0.f;
  for (int d = 0; d < DMODEL; d++) s += line[j * DMODEL + d] * ae[d];
  wa[j] = s;
}

__global__ void k_node_scores(const float* __restrict__ hp, const float* __restrict__ asrc,
                              const float* __restrict__ adst, float* __restrict__ ssrc,
                              float* __restrict__ sdst) {
  int wid  = (blockIdx.x * blockDim.x + threadIdx.x) >> 5;
  int lane = threadIdx.x & 31;
  if (wid >= NNODES) return;
  float4 v  = ((const float4*)(hp + (long)wid * DMODEL))[lane];
  float4 a1 = ((const float4*)asrc)[lane];
  float4 a2 = ((const float4*)adst)[lane];
  float s1 = v.x*a1.x + v.y*a1.y + v.z*a1.z + v.w*a1.w;
  float s2 = v.x*a2.x + v.y*a2.y + v.z*a2.z + v.w*a2.w;
  for (int off = 16; off > 0; off >>= 1) {
    s1 += __shfl_down(s1, off);
    s2 += __shfl_down(s2, off);
  }
  if (lane == 0) { ssrc[wid] = s1; sdst[wid] = s2; }
}

__global__ void k_logit1(const float* __restrict__ e2, const float* __restrict__ wa,
                         const int* __restrict__ eidx, const float* __restrict__ ssrc,
                         const float* __restrict__ sdst, float* __restrict__ logit,
                         unsigned* __restrict__ mxkey) {
  int wid  = (blockIdx.x * blockDim.x + threadIdx.x) >> 5;
  int lane = threadIdx.x & 31;
  const int total = NEDGES + NNODES;
  if (wid >= total) return;
  float p = e2[(long)wid * EDIM + lane] * wa[lane];
  for (int off = 16; off > 0; off >>= 1) p += __shfl_down(p, off);
  if (lane == 0) {
    int s  = wid < NEDGES ? eidx[wid]          : wid - NEDGES;
    int dn = wid < NEDGES ? eidx[NEDGES + wid] : wid - NEDGES;
    float lg = ssrc[s] + sdst[dn] + p;
    lg = lg > 0.f ? lg : NEG_SLOPE * lg;
    logit[wid] = lg;
    atomicMax(&mxkey[dn], fmapkey(lg));
  }
}

__global__ void k_pass2(const float* __restrict__ logit, const unsigned* __restrict__ mxkey,
                        const int* __restrict__ eidx, float* __restrict__ exb,
                        float* __restrict__ den) {
  int i = blockIdx.x * blockDim.x + threadIdx.x;
  const int total = NEDGES + NNODES;
  if (i >= total) return;
  int dn = i < NEDGES ? eidx[NEDGES + i] : i - NEDGES;
  float ex = __expf(logit[i] - funmap(mxkey[dn]));
  exb[i] = ex;
  atomicAdd(&den[dn], ex);
}

__global__ void k_agg(const float* __restrict__ exb, const float* __restrict__ den,
                      const int* __restrict__ eidx, const float* __restrict__ hp,
                      float* __restrict__ agg) {
  long idx = (long)blockIdx.x * blockDim.x + threadIdx.x;
  const long total = (long)(NEDGES + NNODES) * DMODEL;
  if (idx >= total) return;
  int i = (int)(idx >> 7), d = (int)(idx & 127);
  int s  = i < NEDGES ? eidx[i]          : i - NEDGES;
  int dn = i < NEDGES ? eidx[NEDGES + i] : i - NEDGES;
  float alpha = exb[i] / (den[dn] + 1e-16f);
  atomicAdd(&agg[(long)dn * DMODEL + d], hp[(long)s * DMODEL + d] * alpha);
}

// hout (bf16, padded to MP rows) = relu(agg + bias); pad rows -> 0
__global__ void k_bias_relu(const float* __restrict__ agg, const float* __restrict__ bias,
                            __bf16* __restrict__ hout) {
  int idx = blockIdx.x * blockDim.x + threadIdx.x;
  if (idx >= MP * DMODEL) return;
  int m = idx >> 7;
  float v = 0.f;
  if (m < NNODES) {
    v = agg[idx] + bias[idx & 127];
    v = v > 0.f ? v : 0.f;
  }
  hout[idx] = f2bf(v);
}

__global__ void k_colsum(const __bf16* __restrict__ h, float* __restrict__ gsum) {
  int d = threadIdx.x;  // 128 threads
  float acc = 0.f;
  for (int n = blockIdx.x; n < NNODES; n += gridDim.x)
    acc += (float)h[(long)n * DMODEL + d];
  atomicAdd(&gsum[d], acc);
}

__global__ void k_readout(const float* __restrict__ gsum, const float* __restrict__ Wr1,
                          const float* __restrict__ br1, const float* __restrict__ Wr2,
                          const float* __restrict__ br2, float* __restrict__ out) {
  __shared__ float hid[64];
  int t = threadIdx.x;  // 64 threads
  const float invN = 1.0f / (float)NNODES;
  float a = br1[t];
  for (int d = 0; d < DMODEL; d++) a += gsum[d] * invN * Wr1[d * 64 + t];
  hid[t] = a > 0.f ? a : 0.f;
  __syncthreads();
  if (t < SDIM) {
    float o = br2[t];
    for (int j = 0; j < 64; j++) o += hid[j] * Wr2[j * SDIM + t];
    out[t] = o;
  }
}

// ---------------- host-side orchestration ----------------
extern "C" void kernel_launch(void* const* d_in, const int* in_sizes, int n_in,
                              void* d_out, int out_size, void* d_ws, size_t ws_size,
                              hipStream_t stream) {
  (void)in_sizes; (void)n_in; (void)out_size; (void)ws_size;
  const float* x      = (const float*)d_in[0];
  const int*   eidx   = (const int*)  d_in[1];
  const int*   eattr  = (const int*)  d_in[2];
  const float* W1     = (const float*)d_in[3];
  const float* b1     = (const float*)d_in[4];
  const float* W2     = (const float*)d_in[5];
  const float* b2     = (const float*)d_in[6];
  const float* rel    = (const float*)d_in[7];
  const float* lin1   = (const float*)d_in[8];
  const float* asrc1  = (const float*)d_in[9];
  const float* adst1  = (const float*)d_in[10];
  const float* aedge1 = (const float*)d_in[11];
  const float* line1  = (const float*)d_in[12];
  const float* bias1  = (const float*)d_in[13];
  const float* lin2   = (const float*)d_in[14];
  const float* asrc2  = (const float*)d_in[15];
  const float* adst2  = (const float*)d_in[16];
  const float* aedge2 = (const float*)d_in[17];
  const float* line2  = (const float*)d_in[18];
  const float* bias2  = (const float*)d_in[19];
  const float* Wr1    = (const float*)d_in[20];
  const float* br1    = (const float*)d_in[21];
  const float* Wr2    = (const float*)d_in[22];
  const float* br2    = (const float*)d_in[23];

  const long Nn = NNODES, Ee = NEDGES;
  char* ws = (char*)d_ws;
  size_t off = 0;
  auto alloc = [&](size_t bytes) { char* p = ws + off; off = (off + bytes + 255) & ~(size_t)255; return p; };

  // regionA: xb (dead after GEMM1) overlaid by hp + agg (layer phase)
  size_t xb_bytes  = (size_t)MP * KP1 * 2;            // 54.5 MB
  size_t hpa_bytes = (size_t)MP * DMODEL * 4 * 2;     // hp + agg = 51.2 MB
  char* regionA = alloc(xb_bytes > hpa_bytes ? xb_bytes : hpa_bytes);
  __bf16* xb  = (__bf16*)regionA;
  float*  hp  = (float*)regionA;
  float*  agg = (float*)(regionA + (size_t)MP * DMODEL * 4);

  // regionB: t1b (dead after GEMM2) overlaid by hb2
  size_t t1_bytes = (size_t)MP * 256 * 2;             // 25.6 MB
  char* regionB = alloc(t1_bytes);
  __bf16* t1b = (__bf16*)regionB;
  __bf16* hb2 = (__bf16*)regionB;

  __bf16* hb     = (__bf16*)alloc((size_t)MP * DMODEL * 2);
  __bf16* W1t    = (__bf16*)alloc((size_t)256 * KP1 * 2);
  __bf16* W2t    = (__bf16*)alloc((size_t)DMODEL * 256 * 2);
  __bf16* lin1t  = (__bf16*)alloc((size_t)DMODEL * DMODEL * 2);
  __bf16* lin2t  = (__bf16*)alloc((size_t)DMODEL * DMODEL * 2);
  float* e2        = (float*)alloc((size_t)(Ee + Nn) * EDIM * 4);
  float* eloop_sum = (float*)alloc((size_t)Nn * EDIM * 4);
  float* cnt       = (float*)alloc((size_t)Nn * 4);
  float* ssrc      = (float*)alloc((size_t)Nn * 4);
  float* sdst      = (float*)alloc((size_t)Nn * 4);
  float* den       = (float*)alloc((size_t)Nn * 4);
  unsigned* mxkey  = (unsigned*)alloc((size_t)Nn * 4);
  float* logit     = (float*)alloc((size_t)(Ee + Nn) * 4);
  float* exb       = (float*)alloc((size_t)(Ee + Nn) * 4);
  float* wa        = (float*)alloc(256);
  float* gsum      = (float*)alloc(512);

  const int T = 256;
  dim3 g1(MP / BM, 256 / BN);       // encoder GEMM1 (N=256)
  dim3 g2(MP / BM, DMODEL / BN);    // 128-wide GEMMs

  // edge embedding + self-loop mean attrs (independent of encoder GEMMs)
  k_zero<<<(int)((Nn * EDIM + T - 1) / T), T, 0, stream>>>(eloop_sum, Nn * EDIM);
  k_zero<<<(int)((Nn + T - 1) / T), T, 0, stream>>>(cnt, Nn);
  k_edge_embed<<<(NEDGES * EDIM + T - 1) / T, T, 0, stream>>>(eattr, eidx, rel, e2, eloop_sum, cnt);
  k_eloop<<<(NNODES * EDIM + T - 1) / T, T, 0, stream>>>(eloop_sum, cnt, e2);

  // cast operands to padded bf16 (one pass; all GEMM tiles become guard-free)
  k_castA <<<dim3((KP1 + T - 1) / T, MP),  T, 0, stream>>>(x, xb, NNODES, DIN, KP1);
  k_castBT<<<dim3((KP1 + T - 1) / T, 256), T, 0, stream>>>(W1, W1t, DIN, 256, KP1);
  k_castBT<<<dim3((256 + T - 1) / T, DMODEL), T, 0, stream>>>(W2, W2t, 256, DMODEL, 256);
  k_castBT<<<dim3(1, DMODEL), DMODEL, 0, stream>>>(lin1, lin1t, DMODEL, DMODEL, DMODEL);
  k_castBT<<<dim3(1, DMODEL), DMODEL, 0, stream>>>(lin2, lin2t, DMODEL, DMODEL, DMODEL);

  // node encoder MLP (WMMA, bf16 chain)
  k_gemm_bf16wmma<true, true, false, true><<<g1, T, 0, stream>>>(
      xb, W1t, b1, nullptr, t1b, 256, KP1);
  k_gemm_bf16wmma<true, false, false, true><<<g2, T, 0, stream>>>(
      t1b, W2t, b2, nullptr, hb, DMODEL, 256);

  auto run_layer = [&](const __bf16* hin, __bf16* hout, const __bf16* lint,
                       const float* asrc, const float* adst, const float* aedge,
                       const float* linew, const float* bias) {
    k_wa<<<1, 32, 0, stream>>>(linew, aedge, wa);
    k_zero<<<(int)(((long)NNODES * DMODEL + T - 1) / T), T, 0, stream>>>(agg, (long)NNODES * DMODEL);
    k_zero<<<(int)((Nn + T - 1) / T), T, 0, stream>>>(den, Nn);
    k_zero<<<(int)((Nn + T - 1) / T), T, 0, stream>>>((float*)mxkey, Nn);
    k_gemm_bf16wmma<false, false, true, false><<<g2, T, 0, stream>>>(
        hin, lint, nullptr, hp, nullptr, DMODEL, DMODEL);
    k_node_scores<<<(NNODES * 32 + T - 1) / T, T, 0, stream>>>(hp, asrc, adst, ssrc, sdst);
    k_logit1<<<((NEDGES + NNODES) * 32 + T - 1) / T, T, 0, stream>>>(e2, wa, eidx, ssrc, sdst, logit, mxkey);
    k_pass2<<<(NEDGES + NNODES + T - 1) / T, T, 0, stream>>>(logit, mxkey, eidx, exb, den);
    long aggw = (long)(NEDGES + NNODES) * DMODEL;
    k_agg<<<(int)((aggw + T - 1) / T), T, 0, stream>>>(exb, den, eidx, hp, agg);
    k_bias_relu<<<(MP * DMODEL + T - 1) / T, T, 0, stream>>>(agg, bias, hout);
  };

  run_layer(hb,  hb2, lin1t, asrc1, adst1, aedge1, line1, bias1);
  run_layer(hb2, hb,  lin2t, asrc2, adst2, aedge2, line2, bias2);

  // global mean pool + readout MLP
  k_zero<<<1, 128, 0, stream>>>(gsum, 128);
  k_colsum<<<256, 128, 0, stream>>>(hb, gsum);
  k_readout<<<1, 64, 0, stream>>>(gsum, Wr1, br1, Wr2, br2, (float*)d_out);
}
